// SelfAttention_18760417149393
// MI455X (gfx1250) — matrix operand order repaired and checked by reading, not compiled, not run on hardware
//
#include <hip/hip_runtime.h>
#include <hip/hip_bf16.h>

// ---------------------------------------------------------------------------
// Types for CDNA5 WMMA (wave32): v_wmma_f32_16x16x32_bf16
// ---------------------------------------------------------------------------
typedef __attribute__((ext_vector_type(16))) __bf16 v16bf;
typedef __attribute__((ext_vector_type(8)))  __bf16 v8bf;   // 16 bytes
typedef __attribute__((ext_vector_type(4)))  __bf16 v4bf;   // 8 bytes
typedef __attribute__((ext_vector_type(8)))  float  v8f;
typedef __attribute__((ext_vector_type(4)))  float  f32x4;
typedef __attribute__((ext_vector_type(4)))  int    i32x4;
typedef __attribute__((ext_vector_type(4)))  unsigned u32x4;
typedef __attribute__((ext_vector_type(8)))  unsigned u32x8;

#define GLOBAL_AS __attribute__((address_space(1)))
#define LDS_AS    __attribute__((address_space(3)))

typedef GLOBAL_AS i32x4* gp128_t;
typedef LDS_AS    i32x4* lp128_t;

#if defined(__has_builtin)
#if __has_builtin(__builtin_amdgcn_global_store_async_from_lds_b128) && \
    __has_builtin(__builtin_amdgcn_s_wait_asynccnt)
#define USE_ASYNC_ST 1
#endif
#endif
#ifndef USE_ASYNC_ST
#define USE_ASYNC_ST 0
#endif

union Frag16 { v16bf v; v8bf h[2]; };

__device__ __forceinline__ v8f wmma_bf16(v16bf a, v16bf b, v8f c) {
    // (neg_a, A, neg_b, B, c_mod, C, reuse_a, reuse_b)
    return __builtin_amdgcn_wmma_f32_16x16x32_bf16(false, a, false, b, (short)0, c,
                                                   false, false);
}

#define S_LEN 2048
#define D_MODEL 1024
#define N_HEADS 16
#define D_K 64
#define M_ROWS 8192   // B * S

// ---------------------------------------------------------------------------
// fp32 -> bf16 conversion (vectorized by 4)
// ---------------------------------------------------------------------------
__global__ __launch_bounds__(256) void cvt_f32_to_bf16(
    const float* __restrict__ in, __bf16* __restrict__ out, int n4) {
    int i = blockIdx.x * 256 + threadIdx.x;
    if (i < n4) {
        f32x4 v = ((const f32x4*)in)[i];
        v4bf o;
        o[0] = (__bf16)v.x; o[1] = (__bf16)v.y;
        o[2] = (__bf16)v.z; o[3] = (__bf16)v.w;
        ((v4bf*)out)[i] = o;
    }
}

// ---------------------------------------------------------------------------
// Generic GEMM: out = A(M,K) * W(N,K)^T + bias   (bf16 in, fp32 accumulate)
// Workgroup: 8 waves, tile 128M x 16N. W tile staged by the Tensor Data
// Mover in 16x256 chunks, double-buffered in LDS (TENSORcnt pipelined).
// mode 0: store bf16 into (B,H,S,dk) layout      (Q, K projections)
// mode 2: store bf16 into (B,H,dk,S) layout      (V transposed)
// mode 3: store fp32 row-major M x N             (final O)
// ---------------------------------------------------------------------------
#define KC 256   // K-chunk per TDM descriptor

__global__ __launch_bounds__(256) void gemm_bf16(
    const __bf16* __restrict__ A, const __bf16* __restrict__ W,
    const float* __restrict__ bias, void* __restrict__ out, int mode) {
    const int K = D_MODEL;
    const int lane = threadIdx.x & 31;
    const int wave = threadIdx.x >> 5;
    const int col  = lane & 15;       // N within tile / B-matrix column
    const int rsel = lane >> 4;       // which 16-lane half
    const int n0 = blockIdx.x * 16;
    const int m0 = blockIdx.y * 128 + wave * 16;

    __shared__ __bf16 lw[2][16 * KC];     // 2 x 8KB W chunks (rows n0..n0+15)

    const unsigned lds0 = (unsigned)(size_t)&lw[0][0];   // LDS byte offset
    const unsigned long long gbase =
        (unsigned long long)(const char*)(W + (size_t)n0 * K);

    // Build D# and issue TENSOR_LOAD_TO_LDS for chunk c (wave-uniform).
    auto issue_tdm = [&](int c) {
        unsigned long long ga = gbase + (unsigned long long)c * (KC * 2);
        u32x4 g0;
        g0[0] = 1u;                                        // count=1, user mode
        g0[1] = lds0 + (unsigned)((c & 1) * (16 * KC * 2)); // lds_addr
        g0[2] = (unsigned)ga;                              // global_addr[31:0]
        g0[3] = (unsigned)((ga >> 32) & 0x01FFFFFFu) | (2u << 30); // addr|type=2
        const unsigned TD0 = 1u << 20, TD1 = 1u << 20;     // tensor dims (no OOB)
        u32x8 g1;
        g1[0] = 1u << 16;                                  // data_size=1 (2 bytes)
        g1[1] = (TD0 & 0xFFFFu) << 16;                     // tensor_dim0 lo
        g1[2] = ((TD0 >> 16) & 0xFFFFu) | ((TD1 & 0xFFFFu) << 16);
        g1[3] = ((TD1 >> 16) & 0xFFFFu) | ((unsigned)KC << 16);   // tile_dim0=256
        g1[4] = 16u;                                       // tile_dim1=16
        g1[5] = (unsigned)K;                               // tensor_dim0_stride lo
        g1[6] = 0u;
        g1[7] = 0u;
        asm volatile("tensor_load_to_lds %0, %1" :: "s"(g0), "s"(g1) : "memory");
    };

    if (wave == 0) issue_tdm(0);

    v8f acc = {0.f, 0.f, 0.f, 0.f, 0.f, 0.f, 0.f, 0.f};
    const __bf16* arow = A + (size_t)(m0 + col) * K;   // A: M row per lane%16

    for (int c = 0; c < K / KC; ++c) {
        if (wave == 0) {
            if (c + 1 < K / KC) {
                issue_tdm(c + 1);
                __builtin_amdgcn_s_wait_tensorcnt(1);   // oldest DMA done
            } else {
                __builtin_amdgcn_s_wait_tensorcnt(0);
            }
        }
        __syncthreads();
        const __bf16* brow = &lw[c & 1][0] + col * KC; // B: W row (column of B)
        const __bf16* ar   = arow + c * KC;
        for (int kk = 0; kk < KC; kk += 32) {
            Frag16 af, bf;
            // A 16x32 bf16: lanes<16 hold K {kk+0..7, kk+16..23}; lanes>=16 +8
            af.h[0] = *(const v8bf*)(ar + kk + rsel * 8);
            af.h[1] = *(const v8bf*)(ar + kk + 16 + rsel * 8);
            // B 32x16 bf16: lanes<16 hold K {kk+0..7, kk+16..23}; lanes>=16 +8
            bf.h[0] = *(const v8bf*)(brow + kk + rsel * 8);
            bf.h[1] = *(const v8bf*)(brow + kk + 16 + rsel * 8);
            acc = wmma_bf16(af.v, bf.v, acc);
        }
        __syncthreads();                                // before buffer reuse
    }

    const float bv = bias[n0 + col];
    if (mode == 3) {
        float* o = (float*)out;
#pragma unroll
        for (int r = 0; r < 8; r++) {
            int m = m0 + r + rsel * 8;
            o[(size_t)m * D_MODEL + n0 + col] = acc[r] + bv;
        }
    } else {
        __bf16* o = (__bf16*)out;
        const int n = n0 + col, h = n >> 6, d = n & 63;
#pragma unroll
        for (int r = 0; r < 8; r++) {
            int m = m0 + r + rsel * 8;
            int b = m >> 11, s = m & (S_LEN - 1);
            size_t idx = (mode == 2)
                ? ((size_t)((b * N_HEADS + h) * D_K + d) * S_LEN + s)   // (B,H,dk,S)
                : ((size_t)((b * N_HEADS + h) * S_LEN + s) * D_K + d);  // (B,H,S,dk)
            o[idx] = (__bf16)(acc[r] + bv);
        }
    }
}

// ---------------------------------------------------------------------------
// Attention: one WG per (b, h, 16 q-rows). Scores strip kept in LDS,
// softmax in LDS, attn_weights written to global exactly once (async
// LDS->global bulk store when available), PV via WMMA.
// ---------------------------------------------------------------------------
__global__ __launch_bounds__(256) void attn_kernel(
    const __bf16* __restrict__ Qb,   // (B,H,S,dk) bf16
    const __bf16* __restrict__ Kb,   // (B,H,S,dk) bf16
    const __bf16* __restrict__ Vt,   // (B,H,dk,S) bf16
    const int*    __restrict__ amask,// (B,S) int32
    float*        __restrict__ attn, // (B,H,S,S) fp32
    __bf16*       __restrict__ ctx)  // (B,S,D) bf16
{
    __shared__ float  sc[16 * S_LEN];      // 128 KB fp32 scores/probs
    __shared__ __bf16 pb[16 * S_LEN];      //  64 KB bf16 probs (A of PV)
    __shared__ float  red[8 * 16];
    __shared__ float  row_max[16];
    __shared__ float  row_sum[16];

    const int wg = blockIdx.x;
    const int qt = wg & (S_LEN / 16 - 1);   // 0..127
    const int bh = wg >> 7;                 // b*H + h
    const int h  = bh & (N_HEADS - 1);
    const int b  = bh >> 4;
    const int q0 = qt * 16;

    const int lane = threadIdx.x & 31;
    const int wave = threadIdx.x >> 5;
    const int col  = lane & 15;
    const int rsel = lane >> 4;

    const __bf16* Qbase = Qb + ((size_t)bh * S_LEN + q0) * D_K;
    const __bf16* Kbase = Kb + (size_t)bh * S_LEN * D_K;
    const __bf16* Vbase = Vt + (size_t)bh * D_K * S_LEN;

    // ---- Q A-fragments (16 x 64 over two 16x32 fragments), held in regs ----
    Frag16 qa0, qa1;
    {
        const __bf16* qrow = Qbase + (size_t)col * D_K;
        qa0.h[0] = *(const v8bf*)(qrow + rsel * 8);
        qa0.h[1] = *(const v8bf*)(qrow + 16 + rsel * 8);
        qa1.h[0] = *(const v8bf*)(qrow + 32 + rsel * 8);
        qa1.h[1] = *(const v8bf*)(qrow + 48 + rsel * 8);
    }

    // ---- Phase 1: scores = scale * Q K^T, masked, into LDS; track row max --
    const float scale = 0.125f;            // 1/sqrt(64)
    const float NEG = -3.0e38f;
    float wm[8];
#pragma unroll
    for (int r = 0; r < 8; r++) wm[r] = NEG;

    for (int j = wave; j < S_LEN / 16; j += 8) {
        const int k0 = j * 16;
        const int kg = k0 + col;
        if (k0 > q0 + 15) {                 // fully causally masked tile
#pragma unroll
            for (int r = 0; r < 8; r++)
                sc[(r + 8 * rsel) * S_LEN + kg] = NEG;
            continue;
        }
        Frag16 kb0, kb1;
        const __bf16* krow = Kbase + (size_t)kg * D_K;   // B column kg
        __builtin_prefetch(krow + 128 * D_K, 0, 1);      // next tile for this wave
        kb0.h[0] = *(const v8bf*)(krow + rsel * 8);
        kb0.h[1] = *(const v8bf*)(krow + 16 + rsel * 8);
        kb1.h[0] = *(const v8bf*)(krow + 32 + rsel * 8);
        kb1.h[1] = *(const v8bf*)(krow + 48 + rsel * 8);

        v8f acc = {0.f, 0.f, 0.f, 0.f, 0.f, 0.f, 0.f, 0.f};
        acc = wmma_bf16(qa0.v, kb0.v, acc);
        acc = wmma_bf16(qa1.v, kb1.v, acc);

        const int am = amask[b * S_LEN + kg];
#pragma unroll
        for (int r = 0; r < 8; r++) {
            const int m = r + 8 * rsel;     // local q row
            float v = acc[r] * scale;
            if (kg > q0 + m || am == 0) v = NEG;
            sc[m * S_LEN + kg] = v;
            wm[r] = fmaxf(wm[r], v);
        }
    }
    // cross-lane row-max within 16-lane halves, then cross-wave via LDS
#pragma unroll
    for (int r = 0; r < 8; r++) {
        float v = wm[r];
        for (int o = 1; o < 16; o <<= 1) v = fmaxf(v, __shfl_xor(v, o, 32));
        if ((lane & 15) == 0) red[wave * 16 + r + 8 * rsel] = v;
    }
    __syncthreads();
    if (threadIdx.x < 16) {
        float v = red[threadIdx.x];
#pragma unroll
        for (int w = 1; w < 8; w++) v = fmaxf(v, red[w * 16 + threadIdx.x]);
        row_max[threadIdx.x] = v;
        row_sum[threadIdx.x] = 0.f;
    }
    __syncthreads();

    // ---- Phase 2: exp + row sums (ds_add_f32), bf16 copy for PV -----------
    {
        const int m = threadIdx.x >> 4;
        const int kb = (threadIdx.x & 15) * 128;
        const float rm = row_max[m];
        float*  srow = sc + m * S_LEN + kb;
        __bf16* prow = pb + m * S_LEN + kb;
        float lsum = 0.f;
        for (int k = 0; k < 128; k += 4) {
            f32x4 s = *(const f32x4*)(srow + k);
            f32x4 p;
            p.x = __expf(s.x - rm); p.y = __expf(s.y - rm);
            p.z = __expf(s.z - rm); p.w = __expf(s.w - rm);
            lsum += p.x + p.y + p.z + p.w;
            *(f32x4*)(srow + k) = p;
            v4bf pv;
            pv[0] = (__bf16)p.x; pv[1] = (__bf16)p.y;
            pv[2] = (__bf16)p.z; pv[3] = (__bf16)p.w;
            *(v4bf*)(prow + k) = pv;
        }
        atomicAdd(&row_sum[m], lsum);       // LDS float atomic (ds_add_f32)
    }
    __syncthreads();

    // ---- Phase 3: normalize strip in LDS, single write of attn_weights ----
    {
        const int m = threadIdx.x >> 4;
        const int kb = (threadIdx.x & 15) * 128;
        const float inv = 1.0f / row_sum[m];
        float* srow = sc + m * S_LEN + kb;
        float* gout = attn + ((size_t)bh * S_LEN + q0 + m) * S_LEN + kb;
#if USE_ASYNC_ST
        for (int k = 0; k < 128; k += 4) {
            f32x4 p = *(const f32x4*)(srow + k);
            p.x *= inv; p.y *= inv; p.z *= inv; p.w *= inv;
            *(f32x4*)(srow + k) = p;        // normalized in LDS
        }
        asm volatile("s_wait_dscnt 0x0" ::: "memory");   // LDS writes visible to async engine
        {
            // AS1 dst (flat global VA == AS1 VA), AS3 src (flat addr[31:0] == LDS offset)
            gp128_t gdst = (gp128_t)(size_t)gout;
            lp128_t lsrc = (lp128_t)(unsigned)(size_t)srow;
#pragma unroll
            for (int i = 0; i < 32; ++i)                 // 128 floats = 32 x b128
                __builtin_amdgcn_global_store_async_from_lds_b128(
                    gdst + i, lsrc + i, 0, 0);
        }
#else
        for (int k = 0; k < 128; k += 4) {
            f32x4 p = *(const f32x4*)(srow + k);
            p.x *= inv; p.y *= inv; p.z *= inv; p.w *= inv;
            *(f32x4*)(gout + k) = p;
        }
#endif
    }

    // ---- Phase 4: ctx = (P @ V) / rowsum; waves 0..3 own 16 columns each ---
    if (wave < 4) {
        const int n0 = wave * 16;
        const __bf16* vrow = Vbase + (size_t)(n0 + col) * S_LEN;  // Vt row = col of B
        int kend = ((q0 + 16 + 31) >> 5) << 5;                    // causal truncation
        if (kend > S_LEN) kend = S_LEN;
        v8f acc = {0.f, 0.f, 0.f, 0.f, 0.f, 0.f, 0.f, 0.f};
        const __bf16* prow = pb + col * S_LEN;                    // A row = q local
        for (int kk = 0; kk < kend; kk += 32) {
            Frag16 af, bf;
            af.h[0] = *(const v8bf*)(prow + kk + rsel * 8);
            af.h[1] = *(const v8bf*)(prow + kk + 16 + rsel * 8);
            bf.h[0] = *(const v8bf*)(vrow + kk + rsel * 8);
            bf.h[1] = *(const v8bf*)(vrow + kk + 16 + rsel * 8);
            acc = wmma_bf16(af.v, bf.v, acc);
        }
#pragma unroll
        for (int r = 0; r < 8; r++) {
            const int m = r + 8 * rsel;
            const float v = acc[r] / row_sum[m];
            ctx[((size_t)b * S_LEN + q0 + m) * D_MODEL + h * D_K + n0 + col] = (__bf16)v;
        }
    }
#if USE_ASYNC_ST
    __builtin_amdgcn_s_wait_asynccnt(0);    // drain async stores before exit
#endif
}

// ---------------------------------------------------------------------------
// Host-side orchestration
// ---------------------------------------------------------------------------
extern "C" void kernel_launch(void* const* d_in, const int* in_sizes, int n_in,
                              void* d_out, int out_size, void* d_ws, size_t ws_size,
                              hipStream_t stream) {
    const float* x     = (const float*)d_in[0];
    const int*   amask = (const int*)  d_in[1];
    const float* Wq    = (const float*)d_in[2];
    const float* bq    = (const float*)d_in[3];
    const float* Wk    = (const float*)d_in[4];
    const float* bk    = (const float*)d_in[5];
    const float* Wv    = (const float*)d_in[6];
    const float* bv    = (const float*)d_in[7];
    const float* Wo    = (const float*)d_in[8];
    const float* bo    = (const float*)d_in[9];

    // workspace layout (bytes)
    char* ws = (char*)d_ws;
    const size_t XB_N = (size_t)M_ROWS * D_MODEL;        // 8,388,608
    const size_t WB_N = (size_t)D_MODEL * D_MODEL;       // 1,048,576
    __bf16* xb  = (__bf16*)(ws);
    __bf16* Wqb = (__bf16*)(ws + 2 * XB_N);
    __bf16* Wkb = (__bf16*)(ws + 2 * XB_N + 2 * WB_N);
    __bf16* Wvb = (__bf16*)(ws + 2 * XB_N + 4 * WB_N);
    __bf16* Wob = (__bf16*)(ws + 2 * XB_N + 6 * WB_N);
    __bf16* Qb  = (__bf16*)(ws + 2 * XB_N + 8 * WB_N);
    __bf16* Kbf = (__bf16*)(ws + 4 * XB_N + 8 * WB_N);
    __bf16* Vt  = (__bf16*)(ws + 6 * XB_N + 8 * WB_N);
    __bf16* ctx = (__bf16*)(ws + 8 * XB_N + 8 * WB_N);

    float* Obuf    = (float*)d_out;
    float* attnBuf = (float*)d_out + XB_N;               // after O (B*S*D fp32)

    // fp32 -> bf16
    cvt_f32_to_bf16<<<(int)(XB_N / 4 / 256), 256, 0, stream>>>(x,  xb,  (int)(XB_N / 4));
    cvt_f32_to_bf16<<<(int)(WB_N / 4 / 256), 256, 0, stream>>>(Wq, Wqb, (int)(WB_N / 4));
    cvt_f32_to_bf16<<<(int)(WB_N / 4 / 256), 256, 0, stream>>>(Wk, Wkb, (int)(WB_N / 4));
    cvt_f32_to_bf16<<<(int)(WB_N / 4 / 256), 256, 0, stream>>>(Wv, Wvb, (int)(WB_N / 4));
    cvt_f32_to_bf16<<<(int)(WB_N / 4 / 256), 256, 0, stream>>>(Wo, Wob, (int)(WB_N / 4));

    // QKV projections (WMMA bf16, TDM-staged W tiles)
    dim3 gg(D_MODEL / 16, M_ROWS / 128);
    gemm_bf16<<<gg, 256, 0, stream>>>(xb, Wqb, bq, Qb,  0);
    gemm_bf16<<<gg, 256, 0, stream>>>(xb, Wkb, bk, Kbf, 0);
    gemm_bf16<<<gg, 256, 0, stream>>>(xb, Wvb, bv, Vt,  2);

    // attention (scores + softmax + PV), writes attn_weights once
    attn_kernel<<<4 * N_HEADS * (S_LEN / 16), 256, 0, stream>>>(
        Qb, Kbf, Vt, amask, attnBuf, ctx);

    // output projection -> fp32 O
    gemm_bf16<<<gg, 256, 0, stream>>>(ctx, Wob, bo, Obuf, 3);
}
